// ConvolutionalRNN_6493990552237
// MI455X (gfx1250) — matrix operand order repaired and checked by reading, not compiled, not run on hardware
//
#include <hip/hip_runtime.h>
#include <stdint.h>

// ---------------------------------------------------------------------------
// Conv-GRU for MI455X (gfx1250, wave32, WMMA), v2.
//
//   T=128, B=16, CIN=COUT=64, L=256, K=3 (pad 1), GATES=192.
//
// Per step: grid = B * L/16 = 256 workgroups of 12 waves (384 thr).
//   phase 1: 12 waves build the 12 shared B-fragments (bf16 32x16, one per
//            (src,tap,kchunk)) into LDS  -> ds path, built once per WG.
//   phase 2: wave m computes M-tile m (16 gates x 16 pos) for BOTH sources as
//            two independent 6-WMMA chains (v_wmma_f32_16x16x32_bf16);
//            A-fragments are prepacked global (L2/WGP$-hot), B from LDS.
//            waves 0-7: r/z gates (gi+gh summed); waves 8-11: i_n / h_n kept
//            separate (r multiplies only h_n). Raw gates -> LDS.
//   phase 3: 384 threads finalize 64x16 GRU outputs.
// Sequential scan = 128 stream-ordered launches; h_{t-1} read from the
// previous output slice (1 MB, L2-resident; 192 MB L2). x_{t+1} tile is
// prefetched (global_prefetch_b8) to overlap the only real DRAM stream.
// ---------------------------------------------------------------------------

#define T_STEPS 128
#define BATCH   16
#define C_IN    64
#define C_OUT   64
#define SEQL    256
#define MT      12   // 192/16 M-tiles
#define TAPS    3

typedef __attribute__((ext_vector_type(16))) __bf16 v16bf;
typedef __attribute__((ext_vector_type(8)))  float  v8f;

static __device__ __forceinline__ uint32_t f2bf_pack(float lo, float hi) {
  // round-to-nearest-even f32 -> bf16, packed pair
  uint32_t ul = __float_as_uint(lo), uh = __float_as_uint(hi);
  uint32_t rl = (ul + 0x7FFFu + ((ul >> 16) & 1u)) >> 16;
  uint32_t rh = (uh + 0x7FFFu + ((uh >> 16) & 1u)) >> 16;
  return (rl & 0xFFFFu) | (rh << 16);
}

static __device__ __forceinline__ float fsig(float x) {
  return 1.0f / (1.0f + __expf(-x));
}
static __device__ __forceinline__ float ftanh_fast(float x) {
  return 2.0f * fsig(2.0f * x) - 1.0f;
}

// ---------------------------------------------------------------------------
// Pack w_ih / w_hh (f32 [192][64][3]) into a 16-bit A-matrix (16x32)
// VGPR-lane layout with contiguous k = 16h + s: per lane 8 dwords = 16 bf16 K-values.
// ws layout: [src(2)][mtile(12)][tap(3)][kc(2)][lane(32)][dword(8)]
// ---------------------------------------------------------------------------
__global__ __launch_bounds__(256)
void pack_weights_kernel(const float* __restrict__ w_ih,
                         const float* __restrict__ w_hh,
                         uint32_t* __restrict__ wsA) {
  int idx = blockIdx.x * blockDim.x + threadIdx.x;
  if (idx >= 2 * MT * TAPS * 2 * 32 * 8) return;
  int d    = idx & 7;
  int lane = (idx >> 3) & 31;
  int kc   = (idx >> 8) & 1;
  int k    = (idx >> 9) % 3;
  int m    = (idx / 1536) % MT;
  int s    = idx / 18432;
  const float* w = s ? w_hh : w_ih;
  int g = m * 16 + (lane & 15);
  // 16-bit A 16x32: lanes 0-15: V0..V7 -> K 0..15
  //                 lanes16-31: V0..V7 -> K 16..31
  int klo = ((lane < 16) ? 0 : 16) + 2 * d;
  int c0 = kc * 32 + klo;
  float f0 = w[(g * C_IN + c0) * TAPS + k];
  float f1 = w[(g * C_IN + c0 + 1) * TAPS + k];
  wsA[idx] = f2bf_pack(f0, f1);
}

// ---------------------------------------------------------------------------
// One GRU timestep. grid = B*(L/16) = 256 WGs, block = 384 (12 wave32's).
// blockIdx.x: [b(4b) | ltile(4b)]
// ---------------------------------------------------------------------------
__global__ __launch_bounds__(384)
void gru_step_kernel(const float* __restrict__ xt,     // [B][CIN][L]
                     const float* __restrict__ xnext,  // x tile for t+1 (prefetch)
                     const float* __restrict__ hprev,  // [B][COUT][L]
                     const uint32_t* __restrict__ wsA,
                     const float* __restrict__ b_ih,   // [192]
                     const float* __restrict__ b_hh,   // [192]
                     float* __restrict__ out_t) {      // [B][COUT][L]
  __shared__ uint32_t s_bf[12 * 32 * 8];  // 12 B-fragments, bf16 pairs (12 KB)
  __shared__ float    s_rz[128 * 16];     // gi+gh, gates 0..127          (8 KB)
  __shared__ float    s_in[64 * 16];      // i_n, gates 128..191          (4 KB)
  __shared__ float    s_hn[64 * 16];      // h_n, gates 128..191          (4 KB)

  const int tid = threadIdx.x;
  const int ln  = tid & 31;
  const int wid = tid >> 5;                 // 0..11
  const int b   = blockIdx.x >> 4;
  const int l0  = (blockIdx.x & 15) * 16;

  const float* xb = xt    + (size_t)b * C_IN  * SEQL;
  const float* hb = hprev + (size_t)b * C_OUT * SEQL;
  float*       ob = out_t + (size_t)b * C_OUT * SEQL;

  // ---- phase 1: build the 12 shared B fragments into LDS ----------------
  // frag id f = ((src*3 + tap)*2 + kc); wave wid builds frag wid.
  {
    const int f  = wid;
    const int s  = f / 6;
    const int k  = (f >> 1) % 3;
    const int kc = f & 1;
    const float* src = s ? hb : xb;
    const int  p  = l0 + (ln & 15) + k - 1;      // conv "same" shift
    const bool ok = (p >= 0) && (p < SEQL);
    // B 32x16 bf16 layout: lanes 0-15 hold K 0..15, lanes 16-31 hold K 16..31
    const int cb = kc * 32 + ((ln < 16) ? 0 : 16);
    const float* col = src + (size_t)cb * SEQL + p;
    uint32_t* dst = &s_bf[(f * 32 + ln) * 8];
#pragma unroll
    for (int d = 0; d < 8; ++d) {
      float f0 = ok ? col[(2 * d) * SEQL] : 0.0f;
      float f1 = ok ? col[(2 * d + 1) * SEQL] : 0.0f;
      dst[d] = f2bf_pack(f0, f1);
    }
  }

  // Prefetch next timestep's x tile for this (b, l-tile) into L2 (speculative,
  // safe even at t=127 where xnext aliases xt).
  if (tid < C_IN) {
    __builtin_prefetch(xnext + (size_t)b * C_IN * SEQL + (size_t)tid * SEQL + l0, 0, 1);
  }

  __syncthreads();

  // ---- phase 2: per-wave M-tile WMMA accumulation -----------------------
  const int rowoff = (ln < 16) ? 0 : 8;  // C/D 16x16: VGPR r = row r / r+8
  const int m = (wid < 8) ? wid : wid;   // M-tile index == wave id (0..11)

  v8f accx, acch;                        // independent chains per source
#pragma unroll
  for (int r = 0; r < 8; ++r) {
    int g = m * 16 + r + rowoff;
    if (wid < 8) {                       // r/z gates: single combined result
      accx[r] = b_ih[g] + b_hh[g];
      acch[r] = 0.0f;
    } else {                             // n gate: keep i_n / h_n separate
      accx[r] = b_ih[g];
      acch[r] = b_hh[g];
    }
  }

#pragma unroll
  for (int f = 0; f < 12; ++f) {
    const int s  = f / 6;
    const int k  = (f >> 1) % 3;
    const int kc = f & 1;
    v16bf bm = *(const v16bf*)&s_bf[(f * 32 + ln) * 8];          // ds_load_b128 x2
    const uint32_t* ap =
        wsA + ((((size_t)(s * MT + m) * TAPS + k) * 2 + kc) * 32 + ln) * 8;
    v16bf am = *(const v16bf*)ap;                                // global b128 x2
    v8f& acc = s ? acch : accx;
    acc = __builtin_amdgcn_wmma_f32_16x16x32_bf16(
        /*neg_a=*/false, am, /*neg_b=*/false, bm,
        /*c_mod=*/(short)0, acc, /*reuse_a=*/false, /*reuse_b=*/false);
  }

  const int col = ln & 15;
  if (wid < 8) {
#pragma unroll
    for (int r = 0; r < 8; ++r) {
      int g = m * 16 + r + rowoff;                 // 0..127
      s_rz[g * 16 + col] = accx[r] + acch[r];
    }
  } else {
#pragma unroll
    for (int r = 0; r < 8; ++r) {
      int gg = (m - 8) * 16 + r + rowoff;          // 0..63 (gate-128 offset)
      s_in[gg * 16 + col] = accx[r];
      s_hn[gg * 16 + col] = acch[r];
    }
  }
  __syncthreads();

  // ---- phase 3: GRU gate math + state update (1024 outputs / 384 thr) ---
#pragma unroll
  for (int it = 0; it < 3; ++it) {
    int e = tid + it * 384;
    if (e < C_OUT * 16) {
      int co = e >> 4, c = e & 15;
      int l  = l0 + c;
      float rr = fsig(s_rz[co * 16 + c]);                    // reset
      float zz = fsig(s_rz[(64 + co) * 16 + c]);             // update
      float nn = ftanh_fast(s_in[co * 16 + c] + rr * s_hn[co * 16 + c]);
      float hp = hb[(size_t)co * SEQL + l];
      ob[(size_t)co * SEQL + l] = nn + zz * (hp - nn);       // torch GRU update
    }
  }
}

// ---------------------------------------------------------------------------
extern "C" void kernel_launch(void* const* d_in, const int* in_sizes, int n_in,
                              void* d_out, int out_size, void* d_ws, size_t ws_size,
                              hipStream_t stream) {
  const float* x    = (const float*)d_in[0];  // [T][B][CIN][L]
  const float* h0   = (const float*)d_in[1];  // [B][COUT][L]
  const float* w_ih = (const float*)d_in[2];  // [192][64][3]
  const float* w_hh = (const float*)d_in[3];  // [192][64][3]
  const float* b_ih = (const float*)d_in[4];  // [192]
  const float* b_hh = (const float*)d_in[5];  // [192]
  float*       out  = (float*)d_out;          // [T][B][COUT][L]
  uint32_t*    wsA  = (uint32_t*)d_ws;        // 147 KB packed bf16 weights

  // Re-pack every call (deterministic; d_ws may be poisoned).
  pack_weights_kernel<<<144, 256, 0, stream>>>(w_ih, w_hh, wsA);

  const size_t SLICE = (size_t)BATCH * C_OUT * SEQL;  // == B*CIN*L too
  for (int t = 0; t < T_STEPS; ++t) {
    const float* xt = x + (size_t)t * SLICE;
    const float* xn = (t + 1 < T_STEPS) ? xt + SLICE : xt;
    const float* hp = (t == 0) ? h0 : out + (size_t)(t - 1) * SLICE;
    float*       ot = out + (size_t)t * SLICE;
    gru_step_kernel<<<dim3(BATCH * (SEQL / 16)), dim3(384), 0, stream>>>(
        xt, xn, hp, wsA, b_ih, b_hh, ot);
  }
}
